// LSTMTagger_54915451846912
// MI455X (gfx1250) — compile-verified, matrix operand order checked by reading, not run on hardware
//
#include <hip/hip_runtime.h>
#include <hip/hip_bf16.h>
#include <cstdint>
#include <cstddef>

// ---------------------------------------------------------------------------
// LSTM tagger for MI455X (gfx1250, wave32, WMMA).
//   H=256, HH=128, G=4*HH=512, NW=2048 words, TC=12 char steps, TS=4 syl steps
// All dense GEMMs run on v_wmma_f32_16x16x32_bf16 with software-pipelined
// (depth-2) fragment prefetch; the batch-1 word-level recurrence runs as a
// persistent kernel with Whh^T resident in LDS (CDNA5: 320KB/WGP).
// ---------------------------------------------------------------------------

#define HHD   128
#define GSZ   512
#define HDIM  256
#define NWORD 2048
#define TCH   12
#define TSY   4
#define CDIM  64
#define WDIM  256
#define WKDIM 768      // WD + 2*H (word-lstm input K)

typedef __bf16 bf16;
typedef __attribute__((ext_vector_type(16))) __bf16 v16bf;
typedef __attribute__((ext_vector_type(8)))  __bf16 bf16x8;
typedef __attribute__((ext_vector_type(8)))  float  v8f;

__device__ __forceinline__ float sigmf(float x) { return 1.0f / (1.0f + expf(-x)); }

// --------------------------- WMMA fragment helpers -------------------------
// A (16x32 bf16, row-major src, row stride `ld` elements).
// ISA layout: lane l holds row m=l&15; k-half kh = (l<16)?0:8;
// elements e=0..7 -> k=kh+e, e=8..15 -> k=16+kh+(e-8): two contiguous 16B runs.
__device__ __forceinline__ v16bf load_a_frag(const bf16* base, int ld) {
  int lane = threadIdx.x & 31;
  int m    = lane & 15;
  int kh   = (lane >> 4) << 3;
  const bf16* p = base + (size_t)m * ld + kh;
  union { v16bf v; bf16x8 h[2]; } u;
  u.h[0] = *(const bf16x8*)(p);
  u.h[1] = *(const bf16x8*)(p + 16);
  return u.v;
}

// B (32x16 bf16) from pre-swizzled pack: [lane][16] contiguous per (kt,nt) slot.
__device__ __forceinline__ v16bf load_b_frag(const bf16* pack) {
  int lane = threadIdx.x & 31;
  const bf16* p = pack + lane * 16;
  union { v16bf v; bf16x8 h[2]; } u;
  u.h[0] = *(const bf16x8*)(p);
  u.h[1] = *(const bf16x8*)(p + 8);
  return u.v;
}

__device__ __forceinline__ v8f wmma_bf16(v16bf a, v16bf b, v8f c) {
  return __builtin_amdgcn_wmma_f32_16x16x32_bf16(false, a, false, b, (short)0, c,
                                                 false, false);
}

// --------------------------- weight pre-swizzle ----------------------------
// Source W is [N][K] row-major f32 (optionally split along K into W1|W2).
// Emit pack[kt][nt][lane][e] (bf16) matching load_b_frag / ISA B layout.
__global__ void k_pack_b2(const float* __restrict__ W1, int K1,
                          const float* __restrict__ W2, int K2,
                          int N, bf16* __restrict__ pack) {
  int K = K1 + K2;
  int KT = K >> 5, NT = N >> 4;
  int total = KT * NT * 32 * 16;
  for (int idx = blockIdx.x * blockDim.x + threadIdx.x; idx < total;
       idx += gridDim.x * blockDim.x) {
    int e    = idx & 15;
    int lane = (idx >> 4) & 31;
    int slot = idx >> 9;            // kt*NT + nt
    int nt   = slot % NT;
    int kt   = slot / NT;
    int n    = nt * 16 + (lane & 15);
    int kh   = (lane >> 4) << 3;
    int k    = kt * 32 + ((e < 8) ? (kh + e) : (16 + kh + (e - 8)));
    float w  = (k < K1) ? W1[(size_t)n * K1 + k] : W2[(size_t)n * K2 + (k - K1)];
    pack[idx] = (bf16)w;
  }
}

__global__ void k_add_bias(const float* __restrict__ a, const float* __restrict__ b,
                           float* __restrict__ out, int n) {
  int i = blockIdx.x * blockDim.x + threadIdx.x;
  if (i < n) out[i] = a[i] + b[i];
}

// --------------------------- embedding gathers -----------------------------
__global__ void k_gather_x(const int* __restrict__ ids, const float* __restrict__ emb,
                           bf16* __restrict__ X, int T, int B, int D, int ldx) {
  int total = T * B * D;
  for (int idx = blockIdx.x * blockDim.x + threadIdx.x; idx < total;
       idx += gridDim.x * blockDim.x) {
    int d = idx % D;
    int b = (idx / D) % B;
    int t = idx / (D * B);
    X[((size_t)t * B + b) * ldx + d] = (bf16)emb[(size_t)ids[b * T + t] * D + d];
  }
}

// h0[dir][b][j] = emb_prefix[feat[b]][dir*128+j]; c = 0
__global__ void k_init_state(const int* __restrict__ feat, const float* __restrict__ embp,
                             bf16* __restrict__ h_bf, float* __restrict__ c, int B) {
  int total = 2 * B * HHD;
  for (int idx = blockIdx.x * blockDim.x + threadIdx.x; idx < total;
       idx += gridDim.x * blockDim.x) {
    int j   = idx & (HHD - 1);
    int b   = (idx >> 7) % B;
    int dir = idx / (HHD * B);
    h_bf[idx] = (bf16)embp[(size_t)feat[b] * HDIM + dir * HHD + j];
    c[idx]    = 0.0f;
  }
}

// --------------------------- fused LSTM step -------------------------------
// gates[16 rows][512] = [x_t | h] (K = 64+128 = 6 k-tiles) @ Bpack, then cell.
// grid: (B/16, 2 dirs), 512 threads = 16 waves; wave w covers n-tiles 2w, 2w+1.
// K loop is fully unrolled with depth-2 fragment prefetch so WMMAs overlap
// the next tile's global_load_b128s (partial s_wait_loadcnt instead of drain).
__global__ void __launch_bounds__(512)
k_lstm_step(const bf16* __restrict__ X, int s, int T,
            const bf16* __restrict__ Bpack,   // [2][6][32][32][16]
            const float* __restrict__ bias,   // [2][512]
            bf16* __restrict__ h_bf,          // [2][B][128]
            float* __restrict__ c_st,         // [2][B][128]
            float* __restrict__ outf,         // [T][B][256]
            bf16* __restrict__ outb,          // [T][B][256]
            int B) {
  constexpr int KTx = CDIM / 32;              // 2 tiles from x
  constexpr int KT  = KTx + HHD / 32;         // + 4 tiles from h = 6

  int dir  = blockIdx.y;
  int tt   = dir ? (T - 1 - s) : s;           // x index == out index
  int b0   = blockIdx.x * 16;
  int wave = threadIdx.x >> 5;
  int lane = threadIdx.x & 31;

  const bf16* bp   = Bpack + (size_t)dir * KT * 32 * 32 * 16;
  const bf16* hrow = h_bf + ((size_t)dir * B + b0) * HHD;
  const bf16* xrow = X + ((size_t)tt * B + b0) * CDIM;

  int nt0 = wave * 2, nt1 = nt0 + 1;
  auto load_a_kt = [&](int kt) -> v16bf {
    return (kt < KTx) ? load_a_frag(xrow + kt * 32, CDIM)
                      : load_a_frag(hrow + (kt - KTx) * 32, HHD);
  };

  v8f acc0 = {}; v8f acc1 = {};
  v16bf a_c  = load_a_kt(0);
  v16bf b0_c = load_b_frag(bp + ((size_t)nt0 * 32) * 16);
  v16bf b1_c = load_b_frag(bp + ((size_t)nt1 * 32) * 16);
#pragma unroll
  for (int kt = 0; kt < KT; ++kt) {
    v16bf a_n = a_c, b0_n = b0_c, b1_n = b1_c;
    if (kt + 1 < KT) {
      a_n  = load_a_kt(kt + 1);
      b0_n = load_b_frag(bp + ((size_t)((kt + 1) * 32 + nt0) * 32) * 16);
      b1_n = load_b_frag(bp + ((size_t)((kt + 1) * 32 + nt1) * 32) * 16);
    }
    acc0 = wmma_bf16(a_c, b0_c, acc0);
    acc1 = wmma_bf16(a_c, b1_c, acc1);
    a_c = a_n; b0_c = b0_n; b1_c = b1_n;
  }

  __shared__ float gbuf[16 * GSZ];            // 32 KB
  {
    int n  = lane & 15;
    int mo = (lane >> 4) << 3;
#pragma unroll
    for (int r = 0; r < 8; ++r) {
      gbuf[(mo + r) * GSZ + nt0 * 16 + n] = acc0[r];
      gbuf[(mo + r) * GSZ + nt1 * 16 + n] = acc1[r];
    }
  }
  __syncthreads();

  const float* bs = bias + dir * GSZ;
  for (int it = threadIdx.x; it < 16 * HHD; it += blockDim.x) {
    int m = it >> 7;
    int j = it & (HHD - 1);
    int b = b0 + m;
    float gi = gbuf[m * GSZ + j]           + bs[j];
    float gf = gbuf[m * GSZ + HHD + j]     + bs[HHD + j];
    float gg = gbuf[m * GSZ + 2 * HHD + j] + bs[2 * HHD + j];
    float go = gbuf[m * GSZ + 3 * HHD + j] + bs[3 * HHD + j];
    size_t ci = ((size_t)dir * B + b) * HHD + j;
    float cn = sigmf(gf) * c_st[ci] + sigmf(gi) * tanhf(gg);
    float h  = sigmf(go) * tanhf(cn);
    c_st[ci] = cn;
    h_bf[ci] = (bf16)h;
    size_t oi = ((size_t)tt * B + b) * HDIM + dir * HHD + j;
    outf[oi] = h;
    outb[oi] = (bf16)h;
  }
}

// --------------------------- attention -------------------------------------
// e[m] = sum_n tanh( (A[m]@Wa^T)[n] + ba[n] ) * wv[n],  A = out (bf16), N=K=256.
// grid.x = M/16, 512 threads = 16 waves; wave w handles n-tile w. K pipelined.
__global__ void __launch_bounds__(512)
k_attn_energy(const bf16* __restrict__ A, const bf16* __restrict__ Bpack,
              const float* __restrict__ ba, const float* __restrict__ wv,
              float* __restrict__ e) {
  constexpr int KT = HDIM / 32;               // 8
  int m0   = blockIdx.x * 16;
  int wave = threadIdx.x >> 5;
  int lane = threadIdx.x & 31;
  const bf16* arow = A + (size_t)m0 * HDIM;

  v8f acc = {};
  v16bf a_c = load_a_frag(arow, HDIM);
  v16bf b_c = load_b_frag(Bpack + ((size_t)wave * 32) * 16);
#pragma unroll
  for (int kt = 0; kt < KT; ++kt) {
    v16bf a_n = a_c, b_n = b_c;
    if (kt + 1 < KT) {
      a_n = load_a_frag(arow + (kt + 1) * 32, HDIM);
      b_n = load_b_frag(Bpack + ((size_t)((kt + 1) * 16 + wave) * 32) * 16);
    }
    acc = wmma_bf16(a_c, b_c, acc);
    a_c = a_n; b_c = b_n;
  }

  int n = wave * 16 + (lane & 15);
  float part[8];
#pragma unroll
  for (int r = 0; r < 8; ++r) part[r] = tanhf(acc[r] + ba[n]) * wv[n];
#pragma unroll
  for (int off = 1; off < 16; off <<= 1)
#pragma unroll
    for (int r = 0; r < 8; ++r) part[r] += __shfl_xor(part[r], off, 32);

  __shared__ float red[16][16];               // [wave][m]
  if (lane == 0)
#pragma unroll
    for (int r = 0; r < 8; ++r) red[wave][r] = part[r];
  if (lane == 16)
#pragma unroll
    for (int r = 0; r < 8; ++r) red[wave][8 + r] = part[r];
  __syncthreads();
  if (threadIdx.x < 16) {
    float s = 0.0f;
#pragma unroll
    for (int w = 0; w < 16; ++w) s += red[w][threadIdx.x];
    e[m0 + threadIdx.x] = s;
  }
}

__global__ void k_attn_softmax(const float* __restrict__ e, float* __restrict__ attn,
                               int T, int B) {
  int b = blockIdx.x * blockDim.x + threadIdx.x;
  if (b >= B) return;
  float mx = -1e30f;
  for (int t = 0; t < T; ++t) mx = fmaxf(mx, e[t * B + b]);
  float s = 0.0f;
  for (int t = 0; t < T; ++t) s += expf(e[t * B + b] - mx);
  float inv = 1.0f / s;
  for (int t = 0; t < T; ++t) attn[t * B + b] = expf(e[t * B + b] - mx) * inv;
}

// ctx[b][h] = sum_t attn[t][b]*out[t][b][h]  ->  Xw[b][col0+h] (bf16)
__global__ void k_attn_ctx(const float* __restrict__ attn, const float* __restrict__ outf,
                           bf16* __restrict__ Xw, int T, int B, int col0) {
  int idx = blockIdx.x * blockDim.x + threadIdx.x;
  if (idx >= B * HDIM) return;
  int h = idx & (HDIM - 1);
  int b = idx >> 8;
  float s = 0.0f;
  for (int t = 0; t < T; ++t)
    s += attn[t * B + b] * outf[((size_t)t * B + b) * HDIM + h];
  Xw[(size_t)b * WKDIM + col0 + h] = (bf16)s;
}

// --------------------------- word-level precompute GEMM --------------------
// G[dir][b][n] = Xw[b] @ Wih^T[dir] + bias[dir][n]; K = KTILES*32 (templated
// so the pipeline fully unrolls), lda = WKDIM.
template <int KTILES>
__global__ void __launch_bounds__(512)
k_gemm_gates(const bf16* __restrict__ A, const bf16* __restrict__ Bpack,
             const float* __restrict__ bias, float* __restrict__ G,
             int B, int lda) {
  int dir  = blockIdx.y;
  int b0   = blockIdx.x * 16;
  int wave = threadIdx.x >> 5;
  int lane = threadIdx.x & 31;
  int nt0 = wave * 2, nt1 = nt0 + 1;
  const bf16* bp   = Bpack + (size_t)dir * KTILES * 32 * 32 * 16;
  const bf16* arow = A + (size_t)b0 * lda;

  v8f acc0 = {}; v8f acc1 = {};
  v16bf a_c  = load_a_frag(arow, lda);
  v16bf b0_c = load_b_frag(bp + ((size_t)nt0 * 32) * 16);
  v16bf b1_c = load_b_frag(bp + ((size_t)nt1 * 32) * 16);
#pragma unroll
  for (int kt = 0; kt < KTILES; ++kt) {
    v16bf a_n = a_c, b0_n = b0_c, b1_n = b1_c;
    if (kt + 1 < KTILES) {
      a_n  = load_a_frag(arow + (kt + 1) * 32, lda);
      b0_n = load_b_frag(bp + ((size_t)((kt + 1) * 32 + nt0) * 32) * 16);
      b1_n = load_b_frag(bp + ((size_t)((kt + 1) * 32 + nt1) * 32) * 16);
    }
    acc0 = wmma_bf16(a_c, b0_c, acc0);
    acc1 = wmma_bf16(a_c, b1_c, acc1);
    a_c = a_n; b0_c = b0_n; b1_c = b1_n;
  }
  int n  = lane & 15;
  int mo = (lane >> 4) << 3;
#pragma unroll
  for (int r = 0; r < 8; ++r) {
    int m = mo + r;
    G[((size_t)dir * B + b0 + m) * GSZ + nt0 * 16 + n] =
        acc0[r] + bias[dir * GSZ + nt0 * 16 + n];
    G[((size_t)dir * B + b0 + m) * GSZ + nt1 * 16 + n] =
        acc1[r] + bias[dir * GSZ + nt1 * 16 + n];
  }
}

// --------------------------- word-level recurrence -------------------------
// Persistent: 1 workgroup per direction, Whh^T (bf16, 128KB) resident in LDS.
// Per step: g[n] = G[dir][t][n] + sum_k Whh[n][k]*h[k]; cell on threads <128.
__global__ void __launch_bounds__(512)
k_word_lstm(const float* __restrict__ Whh,   // [2][512][128] f32
            const float* __restrict__ G,     // [2][B][512]
            float* __restrict__ outw,        // [B][256]
            int B) {
  extern __shared__ uint8_t smem[];
  bf16*  W  = (bf16*)smem;                        // [128][512] k-major
  float* hb = (float*)(smem + GSZ * HHD * sizeof(bf16));
  float* gb = hb + HHD;
  int dir = blockIdx.x;
  int tid = threadIdx.x;
  const float* Ws = Whh + (size_t)dir * GSZ * HHD;
  for (int idx = tid; idx < GSZ * HHD; idx += blockDim.x) {
    int n = idx >> 7, k = idx & (HHD - 1);
    W[k * GSZ + n] = (bf16)Ws[idx];
  }
  if (tid < HHD) hb[tid] = 0.0f;
  float c = 0.0f;
  __syncthreads();
  for (int s = 0; s < B; ++s) {
    int t = dir ? (B - 1 - s) : s;
    float g = G[((size_t)dir * B + t) * GSZ + tid];
#pragma unroll 8
    for (int k = 0; k < HHD; ++k) g += (float)W[k * GSZ + tid] * hb[k];
    gb[tid] = g;
    __syncthreads();
    if (tid < HHD) {
      float gi = gb[tid], gf = gb[HHD + tid], gg = gb[2 * HHD + tid],
            go = gb[3 * HHD + tid];
      c = sigmf(gf) * c + sigmf(gi) * tanhf(gg);
      float h = sigmf(go) * tanhf(c);
      hb[tid] = h;
      outw[(size_t)t * HDIM + dir * HHD + tid] = h;
    }
    __syncthreads();
  }
}

// --------------------------- classifier + log_softmax ----------------------
__global__ void __launch_bounds__(64)
k_classify(const float* __restrict__ outw,
           const float* __restrict__ Wp, const float* __restrict__ bp,
           const float* __restrict__ Wn, const float* __restrict__ bn,
           float* __restrict__ out, int B) {
  __shared__ float lg[64];
  __shared__ float st[4];
  int b = blockIdx.x, j = threadIdx.x;
  const float* x = outw + (size_t)b * HDIM;
  float v = 0.0f;
  if (j < 47) {
    const float* w = Wp + (size_t)j * HDIM;
    float s = bp[j];
    for (int k = 0; k < HDIM; ++k) s += w[k] * x[k];
    v = s;
  } else if (j < 60) {
    const float* w = Wn + (size_t)(j - 47) * HDIM;
    float s = bn[j - 47];
    for (int k = 0; k < HDIM; ++k) s += w[k] * x[k];
    v = s;
  }
  lg[j] = v;
  __syncthreads();
  if (j == 0) {
    float mx = -1e30f;
    for (int i = 0; i < 47; ++i) mx = fmaxf(mx, lg[i]);
    float s = 0.0f;
    for (int i = 0; i < 47; ++i) s += expf(lg[i] - mx);
    st[0] = mx; st[1] = logf(s);
  }
  if (j == 1) {
    float mx = -1e30f;
    for (int i = 47; i < 60; ++i) mx = fmaxf(mx, lg[i]);
    float s = 0.0f;
    for (int i = 47; i < 60; ++i) s += expf(lg[i] - mx);
    st[2] = mx; st[3] = logf(s);
  }
  __syncthreads();
  if (j < 47)      out[(size_t)b * 47 + j] = lg[j] - st[0] - st[1];
  else if (j < 60) out[(size_t)B * 47 + (size_t)b * 13 + (j - 47)] =
                       lg[j] - st[2] - st[3];
}

// ---------------------------------------------------------------------------
static inline int cdiv(int a, int b) { return (a + b - 1) / b; }

extern "C" void kernel_launch(void* const* d_in, const int* in_sizes, int n_in,
                              void* d_out, int out_size, void* d_ws, size_t ws_size,
                              hipStream_t stream) {
  (void)in_sizes; (void)n_in; (void)out_size; (void)ws_size;
  const int*   word_ids   = (const int*)d_in[0];
  const int*   syl_ids    = (const int*)d_in[1];
  const int*   char_ids   = (const int*)d_in[2];
  const int*   feat_ids   = (const int*)d_in[3];
  const float* emb_char   = (const float*)d_in[4];
  const float* emb_syl    = (const float*)d_in[5];
  const float* emb_word   = (const float*)d_in[6];
  const float* emb_prefix = (const float*)d_in[7];
  const float* aW_c  = (const float*)d_in[8];
  const float* ab_c  = (const float*)d_in[9];
  const float* aw_c  = (const float*)d_in[10];
  const float* aW_s  = (const float*)d_in[11];
  const float* ab_s  = (const float*)d_in[12];
  const float* aw_s  = (const float*)d_in[13];
  const float* Wih_c = (const float*)d_in[14];
  const float* Whh_c = (const float*)d_in[15];
  const float* bih_c = (const float*)d_in[16];
  const float* bhh_c = (const float*)d_in[17];
  const float* Wih_s = (const float*)d_in[18];
  const float* Whh_s = (const float*)d_in[19];
  const float* bih_s = (const float*)d_in[20];
  const float* bhh_s = (const float*)d_in[21];
  const float* Wih_w = (const float*)d_in[22];
  const float* Whh_w = (const float*)d_in[23];
  const float* bih_w = (const float*)d_in[24];
  const float* bhh_w = (const float*)d_in[25];
  const float* W_pos = (const float*)d_in[26];
  const float* b_pos = (const float*)d_in[27];
  const float* W_ner = (const float*)d_in[28];
  const float* b_ner = (const float*)d_in[29];

  // workspace carving (256B aligned)
  uint8_t* ws = (uint8_t*)d_ws;
  size_t off = 0;
  auto alloc = [&](size_t bytes) -> void* {
    void* p = ws + off;
    off = (off + bytes + 255) & ~(size_t)255;
    return p;
  };
  const size_t PK6  = (size_t)6 * 32 * 32 * 16;    // pack elems per dir, K=192
  const size_t PK24 = (size_t)24 * 32 * 32 * 16;   // K=768
  const size_t PKA  = (size_t)8 * 16 * 32 * 16;    // attn, K=256, N=256
  bf16*  bp_char = (bf16*)alloc(2 * PK6 * 2);
  bf16*  bp_syl  = (bf16*)alloc(2 * PK6 * 2);
  bf16*  bp_word = (bf16*)alloc(2 * PK24 * 2);
  bf16*  bp_ac   = (bf16*)alloc(PKA * 2);
  bf16*  bp_as   = (bf16*)alloc(PKA * 2);
  float* bias_c  = (float*)alloc(2 * GSZ * 4);
  float* bias_s  = (float*)alloc(2 * GSZ * 4);
  float* bias_w  = (float*)alloc(2 * GSZ * 4);
  bf16*  Xc      = (bf16*)alloc((size_t)TCH * NWORD * CDIM * 2);
  bf16*  Xs      = (bf16*)alloc((size_t)TSY * NWORD * CDIM * 2);
  bf16*  Xw      = (bf16*)alloc((size_t)NWORD * WKDIM * 2);
  bf16*  hbf     = (bf16*)alloc((size_t)2 * NWORD * HHD * 2);
  float* cst     = (float*)alloc((size_t)2 * NWORD * HHD * 4);
  float* outc_f  = (float*)alloc((size_t)TCH * NWORD * HDIM * 4);
  bf16*  outc_b  = (bf16*)alloc((size_t)TCH * NWORD * HDIM * 2);
  float* outs_f  = (float*)alloc((size_t)TSY * NWORD * HDIM * 4);
  bf16*  outs_b  = (bf16*)alloc((size_t)TSY * NWORD * HDIM * 2);
  float* ebuf    = (float*)alloc((size_t)TCH * NWORD * 4);
  float* abuf    = (float*)alloc((size_t)TCH * NWORD * 4);
  float* Gw      = (float*)alloc((size_t)2 * NWORD * GSZ * 4);
  float* outw    = (float*)alloc((size_t)NWORD * HDIM * 4);

  // 1) weight packs + biases
  for (int d = 0; d < 2; ++d) {
    k_pack_b2<<<cdiv((int)PK6, 256), 256, 0, stream>>>(
        Wih_c + (size_t)d * GSZ * CDIM, CDIM, Whh_c + (size_t)d * GSZ * HHD, HHD,
        GSZ, bp_char + d * PK6);
    k_pack_b2<<<cdiv((int)PK6, 256), 256, 0, stream>>>(
        Wih_s + (size_t)d * GSZ * CDIM, CDIM, Whh_s + (size_t)d * GSZ * HHD, HHD,
        GSZ, bp_syl + d * PK6);
    k_pack_b2<<<cdiv((int)PK24, 256), 256, 0, stream>>>(
        Wih_w + (size_t)d * GSZ * WKDIM, WKDIM, nullptr, 0,
        GSZ, bp_word + d * PK24);
  }
  k_pack_b2<<<cdiv((int)PKA, 256), 256, 0, stream>>>(aW_c, HDIM, nullptr, 0, HDIM, bp_ac);
  k_pack_b2<<<cdiv((int)PKA, 256), 256, 0, stream>>>(aW_s, HDIM, nullptr, 0, HDIM, bp_as);
  k_add_bias<<<4, 256, 0, stream>>>(bih_c, bhh_c, bias_c, 2 * GSZ);
  k_add_bias<<<4, 256, 0, stream>>>(bih_s, bhh_s, bias_s, 2 * GSZ);
  k_add_bias<<<4, 256, 0, stream>>>(bih_w, bhh_w, bias_w, 2 * GSZ);

  // 2) embedding gathers
  k_gather_x<<<cdiv(TCH * NWORD * CDIM, 256), 256, 0, stream>>>(
      char_ids, emb_char, Xc, TCH, NWORD, CDIM, CDIM);
  k_gather_x<<<cdiv(TSY * NWORD * CDIM, 256), 256, 0, stream>>>(
      syl_ids, emb_syl, Xs, TSY, NWORD, CDIM, CDIM);
  k_gather_x<<<cdiv(NWORD * WDIM, 256), 256, 0, stream>>>(
      word_ids, emb_word, Xw, 1, NWORD, WDIM, WKDIM);

  dim3 gstep(NWORD / 16, 2);

  // 3) char BiLSTM + attention  -> Xw cols [256,512)
  k_init_state<<<cdiv(2 * NWORD * HHD, 256), 256, 0, stream>>>(
      feat_ids, emb_prefix, hbf, cst, NWORD);
  for (int s = 0; s < TCH; ++s)
    k_lstm_step<<<gstep, 512, 0, stream>>>(Xc, s, TCH, bp_char, bias_c,
                                           hbf, cst, outc_f, outc_b, NWORD);
  k_attn_energy<<<TCH * NWORD / 16, 512, 0, stream>>>(outc_b, bp_ac, ab_c, aw_c, ebuf);
  k_attn_softmax<<<cdiv(NWORD, 256), 256, 0, stream>>>(ebuf, abuf, TCH, NWORD);
  k_attn_ctx<<<cdiv(NWORD * HDIM, 256), 256, 0, stream>>>(abuf, outc_f, Xw, TCH,
                                                          NWORD, WDIM);

  // 4) syl BiLSTM + attention  -> Xw cols [512,768)
  k_init_state<<<cdiv(2 * NWORD * HHD, 256), 256, 0, stream>>>(
      feat_ids, emb_prefix, hbf, cst, NWORD);
  for (int s = 0; s < TSY; ++s)
    k_lstm_step<<<gstep, 512, 0, stream>>>(Xs, s, TSY, bp_syl, bias_s,
                                           hbf, cst, outs_f, outs_b, NWORD);
  k_attn_energy<<<TSY * NWORD / 16, 512, 0, stream>>>(outs_b, bp_as, ab_s, aw_s, ebuf);
  k_attn_softmax<<<cdiv(NWORD, 256), 256, 0, stream>>>(ebuf, abuf, TSY, NWORD);
  k_attn_ctx<<<cdiv(NWORD * HDIM, 256), 256, 0, stream>>>(abuf, outs_f, Xw, TSY,
                                                          NWORD, WDIM + HDIM);

  // 5) word-level: parallel input GEMM, then persistent recurrence
  k_gemm_gates<24><<<gstep, 512, 0, stream>>>(Xw, bp_word, bias_w, Gw, NWORD, WKDIM);
  size_t lds_word = (size_t)GSZ * HHD * sizeof(bf16) + (HHD + GSZ) * sizeof(float);
  k_word_lstm<<<2, 512, lds_word, stream>>>(Whh_w, Gw, outw, NWORD);

  // 6) classifier heads + log_softmax
  k_classify<<<NWORD, 64, 0, stream>>>(outw, W_pos, b_pos, W_ner, b_ner,
                                       (float*)d_out, NWORD);
}
